// APPNPNet_Structural_74577812128607
// MI455X (gfx1250) — matrix operand order repaired and checked by reading, not compiled, not run on hardware
//
#include <hip/hip_runtime.h>
#include <hip/hip_bf16.h>

typedef __attribute__((ext_vector_type(16))) _Float16 v16h;
typedef __attribute__((ext_vector_type(8)))  float    v8f;

#define D_IN   128
#define NEURON 64
#define ATT_F  8
#define NGRAPH 256
#define ALPHA  0.1f

// ---------------------------------------------------------------- utilities

__device__ __forceinline__ void atomicMaxFloat(float* addr, float v) {
    // signed-int max for v>=0, unsigned-int min for v<0; init must be -inf bits
    if (v >= 0.0f) atomicMax((int*)addr, __float_as_int(v));
    else           atomicMin((unsigned int*)addr, __float_as_uint(v));
}

__global__ void fill_f32(float* __restrict__ p, float v, int n) {
    int i = blockIdx.x * blockDim.x + threadIdx.x;
    if (i < n) p[i] = v;
}

// ---------------------------------------------------------------- APPNP

__global__ void edge_degree(const int* __restrict__ dst, float* __restrict__ deg, int e) {
    int i = blockIdx.x * blockDim.x + threadIdx.x;
    if (i < e) atomicAdd(&deg[dst[i]], 1.0f);
}

__global__ void rsqrt_inplace(float* __restrict__ d, int n) {
    int i = blockIdx.x * blockDim.x + threadIdx.x;
    if (i < n) d[i] = rsqrtf(d[i]);
}

// agg[i,:] = dinv[i]^2 * cur[i,:]   (self-loop contribution; fully overwrites agg)
__global__ void self_init(const float* __restrict__ cur, const float* __restrict__ dinv,
                          float* __restrict__ agg, int n) {
    int idx = blockIdx.x * blockDim.x + threadIdx.x;
    if (idx >= n * D_IN) return;
    int i = idx >> 7;
    float di = dinv[i];
    agg[idx] = di * di * cur[idx];
}

// one wave per edge; lane handles features lane, lane+32, lane+64, lane+96
__global__ void edge_scatter(const float* __restrict__ h,
                             const int* __restrict__ src, const int* __restrict__ dst,
                             const float* __restrict__ dinv,
                             float* __restrict__ agg, int e) {
    int lane = threadIdx.x & 31;
    int edge = blockIdx.x * (blockDim.x >> 5) + (threadIdx.x >> 5);
    if (edge >= e) return;
    int s = src[edge], d = dst[edge];
    float nrm = dinv[s] * dinv[d];
    const float* hs = h + (size_t)s * D_IN;
    float* ad = agg + (size_t)d * D_IN;
#pragma unroll
    for (int j = 0; j < 4; ++j) {
        int f = lane + j * 32;
        atomicAdd(ad + f, nrm * hs[f]);
    }
}

// agg = (1-ALPHA)*agg + ALPHA*x   (in place)
__global__ void appnp_finalize(float* __restrict__ agg, const float* __restrict__ x, int n) {
    int idx = blockIdx.x * blockDim.x + threadIdx.x;
    if (idx < n * D_IN) agg[idx] = (1.0f - ALPHA) * agg[idx] + ALPHA * x[idx];
}

// ---------------------------------------------------------------- WMMA B-fragment prep
// Pack nn1_w [128,64] f32 into f16 fragments in the exact wave32 B layout:
// frag(ks,nt), lane L, element e -> B[K = ks*32 + (e>>3)*16 + (L>=16?8:0) + (e&7)][col = nt*16 + (L&15)]
// index = ((ks*4 + nt)*32 + lane)*16 + e
__global__ void pack_bfrag(const float* __restrict__ W, _Float16* __restrict__ Bf) {
    int idx = blockIdx.x * blockDim.x + threadIdx.x;   // 0 .. 8191
    if (idx >= 4 * 4 * 32 * 16) return;
    int e    = idx & 15;
    int lane = (idx >> 4) & 31;
    int frag = idx >> 9;
    int nt   = frag & 3;
    int ks   = frag >> 2;
    int K    = ks * 32 + ((e >> 3) << 4) + ((lane >= 16) ? 8 : 0) + (e & 7);
    int col  = nt * 16 + (lane & 15);
    Bf[idx] = (_Float16)W[K * NEURON + col];
}

// ---------------------------------------------------------------- dense layer via WMMA
// OUT[N,64] = relu(H[N,128] @ W + b). 4 waves/block; wave = N-tile, block = 16-row M-tile.
// K loop = 4 x v_wmma_f32_16x16x32_f16. A loads are 4x b128 per k-step (row clamped,
// no divergence; OOB rows compute garbage that is never stored).
__global__ void __launch_bounds__(128)
gemm_relu_wmma(const float* __restrict__ H, const _Float16* __restrict__ Bf,
               const float* __restrict__ B, float* __restrict__ OUT, int n) {
    int lane    = threadIdx.x & 31;
    int wave    = threadIdx.x >> 5;      // 0..3 -> N-tile
    int laneLo  = lane & 15;
    int halfSel = lane >> 4;             // 0 or 1
    int mbase   = blockIdx.x * 16;
    int nbase   = wave * 16;

    int row = mbase + laneLo;
    int rowC = row < n ? row : (n - 1);                 // clamp: garbage rows never stored
    const float4* hrow4 = (const float4*)(H + (size_t)rowC * D_IN);
    const v16h* bfp = (const v16h*)Bf;

    v8f c = {};
#pragma unroll
    for (int ks = 0; ks < 4; ++ks) {
        // A (16x32 f16): lane needs K = {abase..abase+7, abase+16..abase+23},
        // abase = ks*32 + halfSel*8  ->  float4 indices base4, +1, +4, +5
        int base4 = ks * 8 + halfSel * 2;
        float av[16];
        *(float4*)&av[0]  = hrow4[base4];
        *(float4*)&av[4]  = hrow4[base4 + 1];
        *(float4*)&av[8]  = hrow4[base4 + 4];
        *(float4*)&av[12] = hrow4[base4 + 5];
        v16h a;
#pragma unroll
        for (int e2 = 0; e2 < 16; ++e2) a[e2] = (_Float16)av[e2];

        v16h b = bfp[(ks * 4 + wave) * 32 + lane];      // pre-packed fragment, 32B/lane

        c = __builtin_amdgcn_wmma_f32_16x16x32_f16(false, a, false, b,
                                                   (short)0, c, false, false);
    }

    int col = nbase + laneLo;
    float bias = B[col];
#pragma unroll
    for (int r = 0; r < 8; ++r) {
        int m = r + halfSel * 8;             // C/D layout: VGPR r -> M=r (+8 for upper half)
        int orow = mbase + m;
        if (orow < n)
            OUT[(size_t)orow * NEURON + col] = fmaxf(c[r] + bias, 0.0f);
    }
}

// ---------------------------------------------------------------- segment softmax + pooling

__global__ void logits_and_max(const float* __restrict__ cf, const float* __restrict__ cw,
                               const float* __restrict__ cb, const int* __restrict__ batch,
                               float* __restrict__ logits, float* __restrict__ m, int n) {
    int i = blockIdx.x * blockDim.x + threadIdx.x;
    if (i >= n) return;
    float s = cb[0];
#pragma unroll
    for (int j = 0; j < ATT_F; ++j) s += cf[i * ATT_F + j] * cw[j];
    logits[i] = s;
    atomicMaxFloat(&m[batch[i]], s);
}

__global__ void softmax_sum(const float* __restrict__ logits, const int* __restrict__ batch,
                            const float* __restrict__ m, float* __restrict__ w,
                            float* __restrict__ z, float* __restrict__ cnt, int n) {
    int i = blockIdx.x * blockDim.x + threadIdx.x;
    if (i >= n) return;
    int g = batch[i];
    float e = expf(logits[i] - m[g]);
    w[i] = e;
    atomicAdd(&z[g], e);
    atomicAdd(&cnt[g], 1.0f);
}

__global__ void pool_max(const float* __restrict__ hout, const float* __restrict__ w,
                         const float* __restrict__ z, const float* __restrict__ cnt,
                         const int* __restrict__ batch, float* __restrict__ pooled, int n) {
    int idx = blockIdx.x * blockDim.x + threadIdx.x;
    if (idx >= n * NEURON) return;
    int i = idx >> 6, f = idx & 63;
    int g = batch[i];
    float p = (w[i] / z[g]) * cnt[g];
    atomicMaxFloat(&pooled[g * NEURON + f], p * hout[idx]);
}

// ---------------------------------------------------------------- final tiny MLP (G=256)

__global__ void final_mlp(const float* __restrict__ pooled,
                          const float* __restrict__ w1, const float* __restrict__ b1,
                          const float* __restrict__ w2, const float* __restrict__ b2,
                          float* __restrict__ out, int g_) {
    __shared__ float sw1[NEURON * 16];
    __shared__ float sb1[16];
    __shared__ float sw2[16];
    for (int i = threadIdx.x; i < NEURON * 16; i += blockDim.x) sw1[i] = w1[i];
    if (threadIdx.x < 16) { sb1[threadIdx.x] = b1[threadIdx.x]; sw2[threadIdx.x] = w2[threadIdx.x]; }
    __syncthreads();
    int g = blockIdx.x * blockDim.x + threadIdx.x;
    if (g >= g_) return;
    const float* pg = pooled + g * NEURON;
    float acc = b2[0];
#pragma unroll 4
    for (int j = 0; j < 16; ++j) {
        float a = sb1[j];
        for (int k = 0; k < NEURON; ++k) a += pg[k] * sw1[k * 16 + j];
        acc += fmaxf(a, 0.0f) * sw2[j];
    }
    out[g] = acc;
}

// ---------------------------------------------------------------- launch

extern "C" void kernel_launch(void* const* d_in, const int* in_sizes, int n_in,
                              void* d_out, int out_size, void* d_ws, size_t ws_size,
                              hipStream_t stream) {
    const float* x      = (const float*)d_in[0];
    const float* cf     = (const float*)d_in[1];
    const int*   eidx   = (const int*)  d_in[2];
    const int*   batch  = (const int*)  d_in[3];
    const float* nn1_w  = (const float*)d_in[5];
    const float* nn1_b  = (const float*)d_in[6];
    const float* closew = (const float*)d_in[7];
    const float* closeb = (const float*)d_in[8];
    const float* att1_w = (const float*)d_in[9];
    const float* att1_b = (const float*)d_in[10];
    const float* att2_w = (const float*)d_in[11];
    const float* att2_b = (const float*)d_in[12];
    float* out = (float*)d_out;

    const int n = in_sizes[0] / D_IN;        // 100000
    const int e = in_sizes[2] / 2;           // 1600000
    const int g = NGRAPH;                    // 256 (compile-time; scalar input not host-readable)
    const int* src = eidx;
    const int* dst = eidx + e;

    // workspace carve-up (256B aligned)
    char* ws = (char*)d_ws;
    size_t off = 0;
    auto carve = [&](size_t bytes) { char* p = ws + off; off += (bytes + 255) & ~(size_t)255; return p; };
    float*     dinv   = (float*)carve((size_t)n * 4);
    float*     buf0   = (float*)carve((size_t)n * D_IN * 4);
    float*     buf1   = (float*)carve((size_t)n * D_IN * 4);
    float*     hout   = (float*)carve((size_t)n * NEURON * 4);
    float*     logits = (float*)carve((size_t)n * 4);
    float*     wbuf   = (float*)carve((size_t)n * 4);
    float*     mbuf   = (float*)carve((size_t)g * 4);
    float*     zbuf   = (float*)carve((size_t)g * 4);
    float*     cntb   = (float*)carve((size_t)g * 4);
    float*     pooled = (float*)carve((size_t)g * NEURON * 4);
    _Float16*  bfrag  = (_Float16*)carve((size_t)4 * 4 * 32 * 16 * 2);   // 16 KB
    (void)ws_size;

    const int T = 256;
    auto blk  = [&](int work) { return (work + T - 1) / T; };
    const int nfeat = n * D_IN;

    // degrees -> dinv
    fill_f32<<<blk(n), T, 0, stream>>>(dinv, 1.0f, n);               // self loops
    edge_degree<<<blk(e), T, 0, stream>>>(dst, dinv, e);
    rsqrt_inplace<<<blk(n), T, 0, stream>>>(dinv, n);

    // pack WMMA B fragments once per call (overlaps with APPNP work in stream order)
    pack_bfrag<<<blk(4 * 4 * 32 * 16), T, 0, stream>>>(nn1_w, bfrag);

    // APPNP: K=3 ping-pong; cur starts at x, result lands in buf0
    const float* cur = x;
    float* aggs[3] = { buf0, buf1, buf0 };
    const int edgesPerBlk = T / 32;
    for (int k = 0; k < 3; ++k) {
        float* agg = aggs[k];
        self_init<<<blk(nfeat), T, 0, stream>>>(cur, dinv, agg, n);
        edge_scatter<<<(e + edgesPerBlk - 1) / edgesPerBlk, T, 0, stream>>>(cur, src, dst, dinv, agg, e);
        appnp_finalize<<<blk(nfeat), T, 0, stream>>>(agg, x, n);
        cur = agg;
    }

    // dense layer with WMMA: one 16-row tile per block, 4 waves cover 64 cols
    gemm_relu_wmma<<<(n + 15) / 16, 128, 0, stream>>>(buf0, bfrag, nn1_b, hout, n);

    // segment softmax accumulators
    fill_f32<<<1, T, 0, stream>>>(mbuf, -INFINITY, g);
    fill_f32<<<1, T, 0, stream>>>(zbuf, 0.0f, g);
    fill_f32<<<1, T, 0, stream>>>(cntb, 0.0f, g);
    fill_f32<<<blk(g * NEURON), T, 0, stream>>>(pooled, -INFINITY, g * NEURON);

    logits_and_max<<<blk(n), T, 0, stream>>>(cf, closew, closeb, batch, logits, mbuf, n);
    softmax_sum<<<blk(n), T, 0, stream>>>(logits, batch, mbuf, wbuf, zbuf, cntb, n);
    pool_max<<<blk(n * NEURON), T, 0, stream>>>(hout, wbuf, zbuf, cntb, batch, pooled, n);

    final_mlp<<<1, T, 0, stream>>>(pooled, att1_w, att1_b, att2_w, att2_b, out, g);
}
